// tDPMPN_34969623724768
// MI455X (gfx1250) — matrix operand order repaired and checked by reading, not compiled, not run on hardware
//
#include <hip/hip_runtime.h>

typedef __attribute__((ext_vector_type(16))) __bf16 v16bf;
typedef __attribute__((ext_vector_type(8)))  __bf16 v8bf;
typedef __attribute__((ext_vector_type(8)))  float  v8f;
typedef unsigned int u32x4 __attribute__((ext_vector_type(4)));
typedef int          i32x8 __attribute__((ext_vector_type(8)));
typedef int          i32x4 __attribute__((ext_vector_type(4)));

#define DIM 256
#define DSM 64

// ---------- helpers ----------
static __device__ __forceinline__ unsigned enc_f32(float x) {
  unsigned u = __float_as_uint(x);
  return (u & 0x80000000u) ? ~u : (u | 0x80000000u);
}
static __device__ __forceinline__ float dec_f32(unsigned e) {
  unsigned u = (e & 0x80000000u) ? (e & 0x7FFFFFFFu) : ~e;
  return __uint_as_float(u);
}

// A-fragment (16x32 bf16) for v_wmma_f32_16x16x32_bf16, built from an fp32 row.
// Lane holds row M = lane&15. elems 0..7 = K = kb + half*8 + i (VGPR0-3),
// elems 8..15 = K = kb + 16 + half*8 + i (VGPR4-7).
static __device__ __forceinline__ v16bf load_a_f32(const float* __restrict__ p, int kb, int half) {
  const float* p0 = p + kb + half * 8;
  const float* p1 = p + kb + 16 + half * 8;
  v16bf a;
#pragma unroll
  for (int i = 0; i < 8; ++i) {
    a[i]     = (__bf16)p0[i];
    a[i + 8] = (__bf16)p1[i];
  }
  return a;
}

// B-fragment (32x16 bf16): lane holds column N = n*16 + (lane&15),
// elems 0..7 = K = kb + half*8 + i, elems 8..15 = K = kb + 16 + half*8 + i, row-major [N][K] weight.
template <typename T>
static __device__ __forceinline__ v16bf load_b_w(const T* w, int stride,
                                                 int kb, int n, int l15, int half) {
  const T* q = w + (size_t)(n * 16 + l15) * stride + kb + half * 8;
  v8bf lo = *(const v8bf*)q;
  v8bf hi = *(const v8bf*)(q + 16);
  return __builtin_shufflevector(lo, hi, 0, 1, 2, 3, 4, 5, 6, 7, 8, 9, 10, 11, 12, 13, 14, 15);
}

// TDM: 1-D contiguous tile load Global->LDS. D# per CDNA5 ISA 8.3/8.4:
// group0: count=1 | lds_addr | global_addr[56:0] | type=2
// group1: data_size code, tensor_dim0 = tile_dim0 = n, tensor_dim1 = tile_dim1 = 1.
// This toolchain's builtin is the 6-arg form: (g0, g1, g2, g3, g4, cpol).
static __device__ __forceinline__ void tdm_load_1d(unsigned ldsOff, const void* g,
                                                   int dsCode, unsigned n) {
  unsigned long long ga = (unsigned long long)g;
  u32x4 g0;
  g0[0] = 1u;                                                  // count=1 (valid user D#)
  g0[1] = ldsOff;                                              // lds_addr (bytes)
  g0[2] = (unsigned)(ga & 0xFFFFFFFFu);                        // global_addr[31:0]
  g0[3] = (unsigned)((ga >> 32) & 0x01FFFFFFu) | (2u << 30);   // global_addr[56:32] | type=2
  i32x8 g1;
  g1[0] = dsCode << 16;                        // wg_mask=0, data_size
  g1[1] = (int)((n & 0xFFFFu) << 16);          // tensor_dim0[15:0]
  g1[2] = (int)((n >> 16) & 0xFFFFu) | (1 << 16); // tensor_dim0[31:16], tensor_dim1=1
  g1[3] = (int)((n & 0xFFFFu) << 16);          // tile_dim0 = n
  g1[4] = 1;                                   // tile_dim1 = 1
  g1[5] = (int)n;                              // tensor_dim0_stride[31:0] = n
  g1[6] = 0;
  g1[7] = 0;
  i32x4 gz  = {};
  i32x8 gz8 = {};
  __builtin_amdgcn_tensor_load_to_lds(g0, g1, gz, gz, gz8, 0);
}

// ---------- weight folding: Ml = Wl[:,0:64]@Wp, Nl = Wl[:,64:128]@Wp, etc. ----------
__global__ void k_fold(const float* __restrict__ Wl, const float* __restrict__ Wr,
                       const float* __restrict__ Wp, const float* __restrict__ Wc,
                       __bf16* __restrict__ Ml, __bf16* __restrict__ Nl,
                       __bf16* __restrict__ Mr, __bf16* __restrict__ Nr,
                       __bf16* __restrict__ Wcb) {
  int t = blockIdx.x * blockDim.x + threadIdx.x;
  if (t < 4 * DSM * DIM) {
    int mat = t >> 14, rem = t & 16383, i = rem >> 8, k = rem & 255;
    const float* W = (mat < 2) ? Wl : Wr;
    int off = (mat & 1) ? 64 : 0;
    float s = 0.f;
    for (int j = 0; j < 64; ++j) s += W[i * DIM + off + j] * Wp[j * DIM + k];
    __bf16* O = (mat == 0) ? Ml : (mat == 1) ? Nl : (mat == 2) ? Mr : Nr;
    O[i * DIM + k] = (__bf16)s;
  } else {
    int u = t - 4 * DSM * DIM;
    if (u < DSM * DSM) Wcb[u] = (__bf16)Wc[u];
  }
}

// ---------- per-query small projections qs|qr|qt -> qsm[b][128] ----------
__global__ void k_qproj(const float* __restrict__ qsrc, const float* __restrict__ qrel,
                        const float* __restrict__ qtim, const float* __restrict__ Ws,
                        const float* __restrict__ bs_, const float* __restrict__ Wp,
                        const float* __restrict__ bp, const float* __restrict__ Wt,
                        const float* __restrict__ bt, float* __restrict__ qsm, int Bq) {
  int t = blockIdx.x * blockDim.x + threadIdx.x;
  if (t >= Bq * 128) return;
  int b = t >> 7, o = t & 127;
  float s;
  if (o < 32) {
    s = bs_[o];
    for (int j = 0; j < 128; ++j) s += Ws[o * 128 + j] * qsrc[b * 128 + j];
  } else if (o < 96) {
    int i = o - 32;
    s = bp[i];
    for (int j = 0; j < 256; ++j) s += Wp[i * 256 + j] * qrel[b * 256 + j];
  } else {
    int i = o - 96;
    s = bt[i];
    for (int j = 0; j < 128; ++j) s += Wt[i * 128 + j] * qtim[b * 128 + j];
  }
  qsm[t] = s;
}

// ---------- fold query terms + biases into per-query 64-vectors cl/cr ----------
__global__ void k_qbias(const float* __restrict__ Wl, const float* __restrict__ bl,
                        const float* __restrict__ Wr, const float* __restrict__ br,
                        const float* __restrict__ bp, const float* __restrict__ qsm,
                        float* __restrict__ cl, float* __restrict__ cr, int Bq) {
  int t = blockIdx.x * blockDim.x + threadIdx.x;
  if (t >= 2 * Bq * DSM) return;
  int side = t / (Bq * DSM), rem = t % (Bq * DSM), b = rem >> 6, i = rem & 63;
  const float* W = side ? Wr : Wl;
  float s = side ? br[i] : bl[i];
  const float* q = qsm + b * 128;
  for (int j = 0; j < 32; ++j) s += W[i * DIM + 128 + j] * q[j];        // qs block
  for (int j = 0; j < 64; ++j) s += W[i * DIM + 160 + j] * q[32 + j];   // qr block
  for (int j = 0; j < 32; ++j) s += W[i * DIM + 224 + j] * q[96 + j];   // qt block
  for (int j = 0; j < 64; ++j) s += (W[i * DIM + j] + W[i * DIM + 64 + j]) * bp[j]; // hv,r bias
  (side ? cr : cl)[b * DSM + i] = s;
}

// ---------- init scratch ----------
__global__ void k_init(unsigned* __restrict__ segmax, float* __restrict__ segsum,
                       float* __restrict__ outv, float* __restrict__ egsum,
                       int* __restrict__ h1, int* __restrict__ h2, int* __restrict__ h3,
                       unsigned* __restrict__ prefix, int* __restrict__ krem,
                       float* __restrict__ kth, const int* __restrict__ maxe, int N, int Bq) {
  int i = blockIdx.x * blockDim.x + threadIdx.x;
  if (i < N) { segmax[i] = 0u; segsum[i] = 0.f; outv[i] = 0.f; }
  if (i < Bq * 4096) { h1[i] = 0; h2[i] = 0; }
  if (i < Bq * 256) h3[i] = 0;
  if (i < Bq) { egsum[i] = 0.f; prefix[i] = 0u; krem[i] = maxe[0]; kth[i] = 0.f; }
}

// ---------- main WMMA kernel: per-edge logits ----------
__global__ __launch_bounds__(256) void k_edge_logits(
    const float* __restrict__ mem, const float* __restrict__ rel,
    const int* __restrict__ esrc, const int* __restrict__ edst, const int* __restrict__ eeg,
    const __bf16* __restrict__ Ml, const __bf16* __restrict__ Nl,
    const __bf16* __restrict__ Mr, const __bf16* __restrict__ Nr,
    const __bf16* __restrict__ Wcb, const float* __restrict__ cl, const float* __restrict__ cr,
    const float* __restrict__ bc, float* __restrict__ logits, int nE, int Bq, int stageCLR) {
  __shared__ alignas(32) __bf16 ldsR[8][16 * DSM];   // per-wave 16x64 bf16 relayout buffer
  __shared__ alignas(32) __bf16 ldsWc[DSM * DSM];    // staged bf16 Wc (TDM)
  __shared__ alignas(16) float  ldsCl[64 * DSM];     // staged per-query left bias (TDM)
  __shared__ alignas(16) float  ldsCr[64 * DSM];     // staged per-query right bias (TDM)

  // Stage shared operands once per workgroup via the Tensor Data Mover.
  if (threadIdx.x == 0) {
    tdm_load_1d((unsigned)(size_t)(const void*)ldsWc, Wcb, /*2B*/ 1, DSM * DSM);
    if (stageCLR) {
      tdm_load_1d((unsigned)(size_t)(const void*)ldsCl, cl, /*4B*/ 2, (unsigned)(Bq * DSM));
      tdm_load_1d((unsigned)(size_t)(const void*)ldsCr, cr, /*4B*/ 2, (unsigned)(Bq * DSM));
    }
    __builtin_amdgcn_s_wait_tensorcnt(0);
  }
  __syncthreads();

  const int lane = threadIdx.x & 31;
  const int wid  = threadIdx.x >> 5;
  const int half = lane >> 4;
  const int l15  = lane & 15;
  const long e0 = ((long)blockIdx.x * 8 + wid) * 16;
  if (e0 >= nE) return;

  const long eM = (e0 + l15 < nE) ? (e0 + l15) : (long)nE - 1;
  const int rs = esrc[eM], rd = edst[eM];
  const float* pS = mem + (size_t)rs * DIM;
  const float* pD = mem + (size_t)rd * DIM;
  const float* pR = rel + (size_t)eM * DIM;
  __builtin_prefetch(pS, 0, 0);   // warm L2 for the gathered node rows
  __builtin_prefetch(pD, 0, 0);

  v8f accL[4] = {};
  v8f accR[4] = {};
  for (int c = 0; c < 8; ++c) {
    const int kb = c * 32;
    v16bf aS = load_a_f32(pS, kb, half);
    v16bf aR = load_a_f32(pR, kb, half);
    v16bf aD = load_a_f32(pD, kb, half);
#pragma unroll
    for (int n = 0; n < 4; ++n) {
      v16bf bMl = load_b_w(Ml, DIM, kb, n, l15, half);
      v16bf bNl = load_b_w(Nl, DIM, kb, n, l15, half);
      v16bf bMr = load_b_w(Mr, DIM, kb, n, l15, half);
      v16bf bNr = load_b_w(Nr, DIM, kb, n, l15, half);
      accL[n] = __builtin_amdgcn_wmma_f32_16x16x32_bf16(false, aS, false, bMl, (short)0, accL[n], false, false);
      accL[n] = __builtin_amdgcn_wmma_f32_16x16x32_bf16(false, aR, false, bNl, (short)0, accL[n], false, false);
      accR[n] = __builtin_amdgcn_wmma_f32_16x16x32_bf16(false, aD, false, bMr, (short)0, accR[n], false, false);
      accR[n] = __builtin_amdgcn_wmma_f32_16x16x32_bf16(false, aR, false, bNr, (short)0, accR[n], false, false);
    }
  }

  // per-edge query bias + leaky relu; stage right side into LDS (row-major 16x64 bf16)
  int egv[8];
#pragma unroll
  for (int v = 0; v < 8; ++v) {
    long ev = e0 + half * 8 + v;
    if (ev >= nE) ev = (long)nE - 1;
    egv[v] = eeg[ev];
  }
  auto biasRelu = [&](const auto* CL, const auto* CR) {
#pragma unroll
    for (int n = 0; n < 4; ++n) {
      const int col = n * 16 + l15;
#pragma unroll
      for (int v = 0; v < 8; ++v) {
        float xl = accL[n][v] + CL[egv[v] * DSM + col];
        float xr = accR[n][v] + CR[egv[v] * DSM + col];
        accL[n][v] = xl > 0.f ? xl : 0.01f * xl;
        float rr = xr > 0.f ? xr : 0.01f * xr;
        ldsR[wid][(v + half * 8) * DSM + col] = (__bf16)rr;
      }
    }
  };
  if (stageCLR) biasRelu(ldsCl, ldsCr);
  else          biasRelu(cl, cr);
  asm volatile("s_wait_dscnt 0" ::: "memory");  // D-layout -> A-layout relayout fence (same wave)

  // C = lrelu(Rpre) @ Wc^T via WMMA (A from LDS, B from TDM-staged LDS Wc)
  v8f accC[4] = {};
#pragma unroll
  for (int c2 = 0; c2 < 2; ++c2) {
    const int kb = c2 * 32;
    const __bf16* rw = &ldsR[wid][l15 * DSM + kb + half * 8];
    v8bf lo = *(const v8bf*)rw;
    v8bf hi = *(const v8bf*)(rw + 16);
    v16bf aC = __builtin_shufflevector(lo, hi, 0, 1, 2, 3, 4, 5, 6, 7, 8, 9, 10, 11, 12, 13, 14, 15);
#pragma unroll
    for (int n = 0; n < 4; ++n) {
      v16bf bC = load_b_w(ldsWc, DSM, kb, n, l15, half);
      accC[n] = __builtin_amdgcn_wmma_f32_16x16x32_bf16(false, aC, false, bC, (short)0, accC[n], false, false);
    }
  }

  // logits[m] = sum_n L[m][n] * (C[m][n] + bc[n]); reduce across the 16 lanes of each half
#pragma unroll
  for (int v = 0; v < 8; ++v) {
    float s = 0.f;
#pragma unroll
    for (int n = 0; n < 4; ++n) s += accL[n][v] * (accC[n][v] + bc[n * 16 + l15]);
    s += __shfl_xor(s, 1, 32);
    s += __shfl_xor(s, 2, 32);
    s += __shfl_xor(s, 4, 32);
    s += __shfl_xor(s, 8, 32);
    const long ew = e0 + half * 8 + v;
    if (l15 == v && ew < nE) logits[ew] = s;
  }
}

// ---------- segment softmax ----------
__global__ void k_segmax(const float* __restrict__ logits, const int* __restrict__ esrc,
                         unsigned* __restrict__ segmax, int E) {
  int e = blockIdx.x * blockDim.x + threadIdx.x;
  if (e >= E) return;
  atomicMax(&segmax[esrc[e]], enc_f32(logits[e]));
}

__global__ void k_expsum(const float* __restrict__ logits, const int* __restrict__ esrc,
                         const unsigned* __restrict__ segmax, float* __restrict__ segsum,
                         float* __restrict__ ex, int E) {
  int e = blockIdx.x * blockDim.x + threadIdx.x;
  if (e >= E) return;
  int s = esrc[e];
  float x = expf(logits[e] - dec_f32(segmax[s]));
  ex[e] = x;
  atomicAdd(&segsum[s], x);
}

__global__ void k_smta(const float* __restrict__ ex, const int* __restrict__ esrc,
                       const float* __restrict__ segsum, const float* __restrict__ natt,
                       float* __restrict__ smv, float* __restrict__ tav, int E) {
  int e = blockIdx.x * blockDim.x + threadIdx.x;
  if (e >= E) return;
  int s = esrc[e];
  float sm = ex[e] / (segsum[s] + 1e-12f);
  smv[e] = sm;
  tav[e] = sm * natt[s];
}

// ---------- exact per-query kth-largest via 3-level radix select on float bits ----------
__global__ void k_hist(const float* __restrict__ tav, const int* __restrict__ eeg,
                       int* __restrict__ hist, const unsigned* __restrict__ prefix,
                       const int* __restrict__ krem, int level, int E) {
  int e = blockIdx.x * blockDim.x + threadIdx.x;
  if (e >= E) return;
  int b = eeg[e];
  if (krem[b] < 0) return;
  unsigned u = __float_as_uint(tav[e]);  // tav >= 0 -> raw bits monotone
  if (level == 1) {
    atomicAdd(&hist[b * 4096 + (int)(u >> 20)], 1);
  } else if (level == 2) {
    if ((u >> 20) == prefix[b]) atomicAdd(&hist[b * 4096 + (int)((u >> 8) & 0xFFFu)], 1);
  } else {
    if ((u >> 8) == prefix[b]) atomicAdd(&hist[b * 256 + (int)(u & 0xFFu)], 1);
  }
}

__global__ void k_scan(const int* __restrict__ hist, unsigned* __restrict__ prefix,
                       int* __restrict__ krem, float* __restrict__ kth,
                       int nbins, int level, int Bq) {
  int b = threadIdx.x;
  if (b >= Bq) return;
  int k = krem[b];
  if (k < 0) return;
  const int* h = hist + b * nbins;
  long cum = 0;
  int bsel = -1;
  for (int i = nbins - 1; i >= 0; --i) {
    int c = h[i];
    if (cum + c >= k) { bsel = i; break; }
    cum += c;
  }
  if (bsel < 0) {  // fewer than k candidates: keep everything (reference kth = -1e30)
    kth[b] = -1e30f;
    krem[b] = -1;
    return;
  }
  krem[b] = k - (int)cum;
  if (level == 1)      prefix[b] = (unsigned)bsel;
  else if (level == 2) prefix[b] = (prefix[b] << 12) | (unsigned)bsel;
  else                 kth[b] = __uint_as_float((prefix[b] << 8) | (unsigned)bsel);
}

// ---------- pruned scatter + renorm ----------
__global__ void k_contrib(const float* __restrict__ tav, const float* __restrict__ smv,
                          const int* __restrict__ eeg, const int* __restrict__ edst,
                          const float* __restrict__ kth, float* __restrict__ outv, int E) {
  int e = blockIdx.x * blockDim.x + threadIdx.x;
  if (e >= E) return;
  float ta = tav[e];
  if (ta >= kth[eeg[e]]) atomicAdd(&outv[edst[e]], smv[e] * ta);
}

__global__ void k_egsum(const float* __restrict__ outv, const int* __restrict__ n2eg,
                        float* __restrict__ egsum, int N) {
  int v = blockIdx.x * blockDim.x + threadIdx.x;
  if (v >= N) return;
  atomicAdd(&egsum[n2eg[v]], outv[v]);
}

__global__ void k_final(float* __restrict__ outv, const int* __restrict__ n2eg,
                        const float* __restrict__ egsum, int N) {
  int v = blockIdx.x * blockDim.x + threadIdx.x;
  if (v >= N) return;
  outv[v] = outv[v] / (egsum[n2eg[v]] + 1e-12f);
}

extern "C" void kernel_launch(void* const* d_in, const int* in_sizes, int n_in,
                              void* d_out, int out_size, void* d_ws, size_t ws_size,
                              hipStream_t stream) {
  (void)n_in; (void)out_size; (void)ws_size;
  const float* natt = (const float*)d_in[0];
  const float* mem  = (const float*)d_in[1];
  const float* rel  = (const float*)d_in[2];
  const float* qsrc = (const float*)d_in[3];
  const float* qrel = (const float*)d_in[4];
  const float* qtim = (const float*)d_in[5];
  const int* eeg  = (const int*)d_in[6];
  const int* esrc = (const int*)d_in[7];
  const int* edst = (const int*)d_in[8];
  const int* n2eg = (const int*)d_in[9];
  const float* Wp  = (const float*)d_in[10];
  const float* bp  = (const float*)d_in[11];
  const float* Ws  = (const float*)d_in[12];
  const float* bs_ = (const float*)d_in[13];
  const float* Wt  = (const float*)d_in[14];
  const float* bt  = (const float*)d_in[15];
  const float* Wl  = (const float*)d_in[16];
  const float* bl  = (const float*)d_in[17];
  const float* Wr  = (const float*)d_in[18];
  const float* br  = (const float*)d_in[19];
  const float* Wc  = (const float*)d_in[20];
  const float* bc  = (const float*)d_in[21];
  const int* maxe  = (const int*)d_in[22];

  const int N  = in_sizes[0];
  const int E  = in_sizes[6];
  const int Bq = in_sizes[3] / 128;

  char* w = (char*)d_ws;
  auto take = [&](size_t bytes) -> char* {
    char* p = w;
    w += (bytes + 255) & ~(size_t)255;
    return p;
  };
  __bf16* Ml  = (__bf16*)take((size_t)DSM * DIM * 2);
  __bf16* Nl  = (__bf16*)take((size_t)DSM * DIM * 2);
  __bf16* Mr  = (__bf16*)take((size_t)DSM * DIM * 2);
  __bf16* Nr  = (__bf16*)take((size_t)DSM * DIM * 2);
  __bf16* Wcb = (__bf16*)take((size_t)DSM * DSM * 2);
  float* qsm    = (float*)take((size_t)Bq * 128 * 4);
  float* cl     = (float*)take((size_t)Bq * DSM * 4);
  float* cr     = (float*)take((size_t)Bq * DSM * 4);
  float* logits = (float*)take((size_t)E * 4);
  float* ex     = (float*)take((size_t)E * 4);
  float* smv    = (float*)take((size_t)E * 4);
  float* tav    = (float*)take((size_t)E * 4);
  unsigned* segmax = (unsigned*)take((size_t)N * 4);
  float* segsum    = (float*)take((size_t)N * 4);
  float* egsum     = (float*)take((size_t)Bq * 4);
  int* h1 = (int*)take((size_t)Bq * 4096 * 4);
  int* h2 = (int*)take((size_t)Bq * 4096 * 4);
  int* h3 = (int*)take((size_t)Bq * 256 * 4);
  unsigned* prefix = (unsigned*)take((size_t)Bq * 4);
  int* krem        = (int*)take((size_t)Bq * 4);
  float* kth       = (float*)take((size_t)Bq * 4);
  float* outv = (float*)d_out;

  const int stageCLR = (Bq <= 64) ? 1 : 0;

  int t;
  t = 4 * DSM * DIM + DSM * DSM;
  k_fold<<<(t + 255) / 256, 256, 0, stream>>>(Wl, Wr, Wp, Wc, Ml, Nl, Mr, Nr, Wcb);
  t = Bq * 128;
  k_qproj<<<(t + 255) / 256, 256, 0, stream>>>(qsrc, qrel, qtim, Ws, bs_, Wp, bp, Wt, bt, qsm, Bq);
  t = 2 * Bq * DSM;
  k_qbias<<<(t + 255) / 256, 256, 0, stream>>>(Wl, bl, Wr, br, bp, qsm, cl, cr, Bq);
  t = (N > Bq * 4096) ? N : Bq * 4096;
  k_init<<<(t + 255) / 256, 256, 0, stream>>>(segmax, segsum, outv, egsum, h1, h2, h3,
                                              prefix, krem, kth, maxe, N, Bq);

  k_edge_logits<<<(E + 127) / 128, 256, 0, stream>>>(mem, rel, esrc, edst, eeg, Ml, Nl, Mr, Nr,
                                                     Wcb, cl, cr, bc, logits, E, Bq, stageCLR);

  k_segmax<<<(E + 255) / 256, 256, 0, stream>>>(logits, esrc, segmax, E);
  k_expsum<<<(E + 255) / 256, 256, 0, stream>>>(logits, esrc, segmax, segsum, ex, E);
  k_smta<<<(E + 255) / 256, 256, 0, stream>>>(ex, esrc, segsum, natt, smv, tav, E);

  k_hist<<<(E + 255) / 256, 256, 0, stream>>>(tav, eeg, h1, prefix, krem, 1, E);
  k_scan<<<1, Bq, 0, stream>>>(h1, prefix, krem, kth, 4096, 1, Bq);
  k_hist<<<(E + 255) / 256, 256, 0, stream>>>(tav, eeg, h2, prefix, krem, 2, E);
  k_scan<<<1, Bq, 0, stream>>>(h2, prefix, krem, kth, 4096, 2, Bq);
  k_hist<<<(E + 255) / 256, 256, 0, stream>>>(tav, eeg, h3, prefix, krem, 3, E);
  k_scan<<<1, Bq, 0, stream>>>(h3, prefix, krem, kth, 256, 3, Bq);

  k_contrib<<<(E + 255) / 256, 256, 0, stream>>>(tav, smv, eeg, edst, kth, outv, E);
  k_egsum<<<(N + 255) / 256, 256, 0, stream>>>(outv, n2eg, egsum, N);
  k_final<<<(N + 255) / 256, 256, 0, stream>>>(outv, n2eg, egsum, N);
}
